// LSTM_35467839931010
// MI455X (gfx1250) — matrix operand order repaired and checked by reading, not compiled, not run on hardware
//
#include <hip/hip_runtime.h>

// ---------------- problem constants ----------------
constexpr int Lc = 2;
constexpr int Bc = 64;
constexpr int Tc = 512;
constexpr int Hc = 1024;
constexpr int FOURH = 4 * Hc;   // 4096
constexpr int TWOH  = 2 * Hc;   // 2048
constexpr size_t BTH = (size_t)Bc * Tc * Hc;   // 33,554,432
constexpr size_t BH  = (size_t)Bc * Hc;        // 65,536
constexpr size_t LBH = (size_t)Lc * BH;        // 131,072
constexpr size_t WEL = (size_t)Lc * FOURH * TWOH; // 16,777,216

// A slab in LDS: 64 rows x 1024 halves per phase, padded pitch for bank spread
constexpr int APITCH = 1032;                   // halves; 2064 B rows (16B aligned)

// ---------------- vector types ----------------
typedef __attribute__((ext_vector_type(8)))  __bf16 v8bf;
typedef __attribute__((ext_vector_type(16))) __bf16 v16bf;
typedef __attribute__((ext_vector_type(8)))  float  v8f;

union V16U { v16bf v; v8bf h[2]; };

// fp32 -> bf16 (round-to-nearest-even), raw bits
__device__ __forceinline__ unsigned short f2bf(float f) {
  unsigned u = __builtin_bit_cast(unsigned, f);
  u += 0x7FFFu + ((u >> 16) & 1u);
  return (unsigned short)(u >> 16);
}

__device__ __forceinline__ float fsigmoid(float x) {
  return 1.0f / (1.0f + __expf(-x));
}
__device__ __forceinline__ float ftanh(float x) {
  return 1.0f - 2.0f / (__expf(2.0f * x) + 1.0f);
}

// LDS byte-address of a generic pointer into a __shared__ array
typedef __attribute__((address_space(3))) unsigned short lds_ushort;
__device__ __forceinline__ unsigned lds_addr(unsigned short* p) {
  return (unsigned)(uintptr_t)(lds_ushort*)p;
}

// CDNA5 async copy: global -> LDS, 16 bytes per lane, tracked by ASYNCcnt
__device__ __forceinline__ void async_ld_b128(unsigned lds_off, const unsigned short* gptr) {
  asm volatile("global_load_async_to_lds_b128 %0, %1, off"
               :: "v"(lds_off), "v"((unsigned long long)(uintptr_t)gptr)
               : "memory");
}
__device__ __forceinline__ void wait_async0() {
  asm volatile("s_wait_asynccnt 0x0" ::: "memory");
}

// ---------------- conversion: f32 -> bf16 (8 elems / thread) ----------------
__global__ __launch_bounds__(256) void cvt_f32_bf16(const float* __restrict__ in,
                                                    unsigned short* __restrict__ out,
                                                    size_t n) {
  size_t i = ((size_t)blockIdx.x * blockDim.x + threadIdx.x) * 8;
  if (i >= n) return;
  const float4 a = reinterpret_cast<const float4*>(in + i)[0];
  const float4 b = reinterpret_cast<const float4*>(in + i)[1];
  uint4 o;
  o.x = (unsigned)f2bf(a.x) | ((unsigned)f2bf(a.y) << 16);
  o.y = (unsigned)f2bf(a.z) | ((unsigned)f2bf(a.w) << 16);
  o.z = (unsigned)f2bf(b.x) | ((unsigned)f2bf(b.y) << 16);
  o.w = (unsigned)f2bf(b.z) | ((unsigned)f2bf(b.w) << 16);
  *reinterpret_cast<uint4*>(out + i) = o;
}

// ---------------- state init ----------------
__global__ __launch_bounds__(256) void init_state(const float* __restrict__ h0,
                                                  const float* __restrict__ c0,
                                                  unsigned short* __restrict__ hbf, // [L][2][B*H]
                                                  float* __restrict__ hf,           // [L][B*H]
                                                  float* __restrict__ cf) {         // [L][B*H]
  size_t idx = (size_t)blockIdx.x * blockDim.x + threadIdx.x;
  if (idx >= LBH) return;
  size_t l = idx / BH, r = idx % BH;
  float h = h0[idx];
  hf[idx] = h;
  cf[idx] = c0[idx];
  hbf[l * 2 * BH + r] = f2bf(h);  // buffer 0 is the t=0 read buffer
}

// ---------------- finalize: h_n / c_n -> d_out tail ----------------
__global__ __launch_bounds__(256) void finalize_state(const float* __restrict__ hf,
                                                      const float* __restrict__ cf,
                                                      float* __restrict__ out) {
  size_t idx = (size_t)blockIdx.x * blockDim.x + threadIdx.x;
  if (idx >= LBH) return;
  out[idx]       = hf[idx];
  out[LBH + idx] = cf[idx];
}

// ---------------- fused LSTM step ----------------
// grid = 32 WGs (WG owns hidden cols [32*wg, 32*wg+32)), block = 256 = 8 waves.
// wave w -> gate g = w/2, n-subtile = w%2; each wave: 4 M-tiles x 64 K-iters of WMMA.
// Phase p stages A[:, p*1024 : (p+1)*1024] (x_t slab, then h slab) into LDS via
// async global->LDS copies; each phase then runs 32 barrier-free K iterations.
__global__ __launch_bounds__(256) void lstm_step(
    const unsigned short* __restrict__ Ain,   // [B,T,H] bf16 layer input
    const unsigned short* __restrict__ Wl,    // [4H,2H] bf16 row-major
    const float* __restrict__ bl,             // [4H]
    const unsigned short* __restrict__ hread, // [B,H] bf16 (t-1 state)
    unsigned short* __restrict__ hwrite,      // [B,H] bf16 (t state)
    float* __restrict__ cbuf,                 // [B,H] f32 cell state (rw)
    float* __restrict__ hf32,                 // [B,H] f32 h state (w)
    unsigned short* __restrict__ seq_bf,      // [B,T,H] bf16 out or null
    float* __restrict__ seq_f32,              // [B,T,H] f32 out or null
    int t) {
  // 64 x 1032 halves = 132,096 B; gate-exchange buffer aliased on top afterwards
  __shared__ __align__(16) unsigned short Smem[64 * APITCH];
  unsigned short* Ald = Smem;
  auto Gld = reinterpret_cast<float (*)[64][36]>(&Smem[0]);  // [4][64][36] = 36,864 B

  const int tid  = threadIdx.x;
  const int lane = tid & 31;
  const int wave = tid >> 5;
  const int gate = wave >> 1;
  const int nt   = wave & 1;
  const int j0   = blockIdx.x * 32;
  const int ln16 = lane & 15;
  const int ncol  = gate * Hc + j0 + nt * 16 + ln16;  // output/gate column in [0,4H)
  const int koffB = (lane < 16) ? 0 : 16;             // B-frag: K 0..15 / 16..31
  const int kcA   = (lane >> 4) * 16;                 // A-frag: K 0..15 / 16..31

  const unsigned short* Wrow = Wl + (size_t)ncol * TWOH;

  v8f acc[4] = {};  // 4 M-tiles (M = 0,16,32,48)

  for (int ph = 0; ph < 2; ++ph) {
    // ---- bulk async stage: 64 rows x 1024 halves (128 KiB), 32 x b128 per thread ----
    const unsigned short* src = ph ? hread : Ain;
    const size_t rowPitch = ph ? (size_t)Hc : (size_t)Tc * Hc;
    const size_t srcOfs   = ph ? 0 : (size_t)t * Hc;
#pragma unroll 4
    for (int i = 0; i < 32; ++i) {
      const int idx = i * 256 + tid;
      const int row = idx >> 7;            // 128 x 16B chunks per row
      const int c8  = (idx & 127) << 3;    // half-index within row
      async_ld_b128(lds_addr(&Ald[row * APITCH + c8]),
                    src + (size_t)row * rowPitch + srcOfs + c8);
    }
    wait_async0();
    __syncthreads();

    // ---- 32 barrier-free K iterations over this phase ----
    const int kbase = ph ? Hc : 0;
    for (int kk = 0; kk < 32; ++kk) {
      const int kl = kk * 32;
      const v16bf bfrag =
          *reinterpret_cast<const v16bf*>(Wrow + kbase + kl + koffB);
#pragma unroll
      for (int mt = 0; mt < 4; ++mt) {
        unsigned short* ap = &Ald[(mt * 16 + ln16) * APITCH + kl + kcA];
        V16U af;
        af.h[0] = *reinterpret_cast<const v8bf*>(ap);       // K {0..7} / {16..23}
        af.h[1] = *reinterpret_cast<const v8bf*>(ap + 8);   // K {8..15} / {24..31}
        acc[mt] = __builtin_amdgcn_wmma_f32_16x16x32_bf16(
            false, af.v, false, bfrag, (short)0, acc[mt], false, false);
      }
    }
    __syncthreads();  // all waves done with this slab before overwrite / G aliasing
  }

  // ---- bias + scatter gates to LDS (aliases A slab; safe after barrier) ----
  const float bv  = bl[ncol];
  const int jj    = nt * 16 + ln16;
  const int mbase = (lane >> 4) * 8;  // C/D layout: VGPR r -> M = r (+8 for hi lanes)
#pragma unroll
  for (int mt = 0; mt < 4; ++mt) {
#pragma unroll
    for (int r = 0; r < 8; ++r) {
      Gld[gate][mt * 16 + mbase + r][jj] = acc[mt][r] + bv;
    }
  }
  __syncthreads();

  // ---- fused cell update: thread owns (m = tid/4, 8 consecutive cols) ----
  {
    const int um = tid >> 2;           // 0..63
    const int uj = (tid & 3) << 3;     // 0,8,16,24
    const int j  = j0 + uj;
    const size_t sidx = (size_t)um * Hc + j;
    const size_t qidx = ((size_t)um * Tc + t) * Hc + j;

    float fg[8], ig[8], gg[8], og[8], co[8], cn[8], hn[8];
    *reinterpret_cast<float4*>(&fg[0]) = *reinterpret_cast<const float4*>(&Gld[0][um][uj]);
    *reinterpret_cast<float4*>(&fg[4]) = *reinterpret_cast<const float4*>(&Gld[0][um][uj + 4]);
    *reinterpret_cast<float4*>(&ig[0]) = *reinterpret_cast<const float4*>(&Gld[1][um][uj]);
    *reinterpret_cast<float4*>(&ig[4]) = *reinterpret_cast<const float4*>(&Gld[1][um][uj + 4]);
    *reinterpret_cast<float4*>(&gg[0]) = *reinterpret_cast<const float4*>(&Gld[2][um][uj]);
    *reinterpret_cast<float4*>(&gg[4]) = *reinterpret_cast<const float4*>(&Gld[2][um][uj + 4]);
    *reinterpret_cast<float4*>(&og[0]) = *reinterpret_cast<const float4*>(&Gld[3][um][uj]);
    *reinterpret_cast<float4*>(&og[4]) = *reinterpret_cast<const float4*>(&Gld[3][um][uj + 4]);
    *reinterpret_cast<float4*>(&co[0]) = *reinterpret_cast<const float4*>(cbuf + sidx);
    *reinterpret_cast<float4*>(&co[4]) = *reinterpret_cast<const float4*>(cbuf + sidx + 4);

#pragma unroll
    for (int r = 0; r < 8; ++r) {
      cn[r] = fsigmoid(fg[r]) * co[r] + fsigmoid(ig[r]) * ftanh(gg[r]);
      hn[r] = fsigmoid(og[r]) * ftanh(cn[r]);
    }

    *reinterpret_cast<float4*>(cbuf + sidx)     = *reinterpret_cast<float4*>(&cn[0]);
    *reinterpret_cast<float4*>(cbuf + sidx + 4) = *reinterpret_cast<float4*>(&cn[4]);
    *reinterpret_cast<float4*>(hf32 + sidx)     = *reinterpret_cast<float4*>(&hn[0]);
    *reinterpret_cast<float4*>(hf32 + sidx + 4) = *reinterpret_cast<float4*>(&hn[4]);

    uint4 hb;
    hb.x = (unsigned)f2bf(hn[0]) | ((unsigned)f2bf(hn[1]) << 16);
    hb.y = (unsigned)f2bf(hn[2]) | ((unsigned)f2bf(hn[3]) << 16);
    hb.z = (unsigned)f2bf(hn[4]) | ((unsigned)f2bf(hn[5]) << 16);
    hb.w = (unsigned)f2bf(hn[6]) | ((unsigned)f2bf(hn[7]) << 16);
    *reinterpret_cast<uint4*>(hwrite + sidx) = hb;

    if (seq_bf)  *reinterpret_cast<uint4*>(seq_bf + qidx) = hb;
    if (seq_f32) {
      *reinterpret_cast<float4*>(seq_f32 + qidx)     = *reinterpret_cast<float4*>(&hn[0]);
      *reinterpret_cast<float4*>(seq_f32 + qidx + 4) = *reinterpret_cast<float4*>(&hn[4]);
    }
  }
}

// ---------------- host driver ----------------
extern "C" void kernel_launch(void* const* d_in, const int* in_sizes, int n_in,
                              void* d_out, int out_size, void* d_ws, size_t ws_size,
                              hipStream_t stream) {
  const float* x  = (const float*)d_in[0];  // [B,T,H]
  const float* W  = (const float*)d_in[1];  // [L,4H,2H]
  const float* b  = (const float*)d_in[2];  // [L,4H]
  const float* h0 = (const float*)d_in[3];  // [L,B,H]
  const float* c0 = (const float*)d_in[4];  // [L,B,H]
  float* out = (float*)d_out;               // [B,T,H] ++ [L,B,H] ++ [L,B,H]

  // workspace carve-up (bf16 stored as ushort)
  char* ws = (char*)d_ws;
  unsigned short* Wbf   = (unsigned short*)ws;                    // WEL halves
  unsigned short* xbf   = Wbf + WEL;                              // BTH halves
  unsigned short* midbf = xbf + BTH;                              // BTH halves
  unsigned short* hbf   = midbf + BTH;                            // L*2*BH halves (double-buffered h)
  float* hf = (float*)(hbf + (size_t)Lc * 2 * BH);                // LBH floats
  float* cf = hf + LBH;                                           // LBH floats

  cvt_f32_bf16<<<(unsigned)(WEL / 8 / 256), 256, 0, stream>>>(W, Wbf, WEL);
  cvt_f32_bf16<<<(unsigned)(BTH / 8 / 256), 256, 0, stream>>>(x, xbf, BTH);
  init_state<<<(unsigned)(LBH / 256), 256, 0, stream>>>(h0, c0, hbf, hf, cf);

  for (int l = 0; l < Lc; ++l) {
    const unsigned short* Ain = (l == 0) ? xbf : midbf;
    const unsigned short* Wl  = Wbf + (size_t)l * FOURH * TWOH;
    const float* bl = b + (size_t)l * FOURH;
    unsigned short* hbfl = hbf + (size_t)l * 2 * BH;
    float* cl = cf + (size_t)l * BH;
    float* hl = hf + (size_t)l * BH;
    unsigned short* seqb = (l == 0) ? midbf : nullptr;  // feed next layer (bf16)
    float* seqf = (l == Lc - 1) ? out : nullptr;        // final f32 output
    for (int t = 0; t < Tc; ++t) {
      lstm_step<<<32, 256, 0, stream>>>(
          Ain, Wl, bl,
          hbfl + (size_t)(t & 1) * BH,        // read h_{t-1}
          hbfl + (size_t)((t + 1) & 1) * BH,  // write h_t
          cl, hl, seqb, seqf, t);
    }
  }

  finalize_state<<<(unsigned)(LBH / 256), 256, 0, stream>>>(hf, cf, out + BTH);
}
